// EnhancedBaselineWithReturnBoost_85933705658779
// MI455X (gfx1250) — matrix operand order repaired and checked by reading, not compiled, not run on hardware
//
#include <hip/hip_runtime.h>
#include <hip/hip_bf16.h>
#include <math.h>

// Problem constants (from reference)
#define VOC   50000
#define D     256
#define DI    1024
#define NL    4
#define NH    8
#define BSZ   256
#define SEQ   128
#define DK    32
#define BS    (BSZ * SEQ)   // 32768 tokens

typedef _Float16 h8   __attribute__((ext_vector_type(8)));
typedef _Float16 v16h __attribute__((ext_vector_type(16)));
typedef float    v8f  __attribute__((ext_vector_type(8)));

// ---------------------------------------------------------------------------
// WMMA fragment helpers (CDNA5 wave32 layouts, 16x16x32 f16 -> f32)
// A (16x32, row-major source): lane l<16 holds row M=l with K = {0..7, 16..23},
// lanes 16..31 hold same rows with K = {8..15, 24..31}.
// ---------------------------------------------------------------------------
__device__ __forceinline__ v16h load_a_frag(const _Float16* __restrict__ A,
                                            int lda, int row0, int k0) {
    int lane = threadIdx.x & 31;
    int l = lane & 15, hi = lane >> 4;
    const _Float16* p = A + (size_t)(row0 + l) * lda + k0 + hi * 8;
    h8 lo = *(const h8*)(p);
    h8 hv = *(const h8*)(p + 16);
    v16h r;
#pragma unroll
    for (int i = 0; i < 8; ++i) { r[i] = lo[i]; r[8 + i] = hv[i]; }
    return r;
}

// B (32x16) loaded from a K-major ("Bt") buffer: Bt[n][k], contiguous in k.
// lane = column n (0..15) with K = {0..7, 16..23}; lanes 16..31 hold K = {8..15, 24..31}.
__device__ __forceinline__ v16h load_bt_frag(const _Float16* __restrict__ Bt,
                                             int ldk, int col0, int k0) {
    int lane = threadIdx.x & 31;
    int n = lane & 15, hi = lane >> 4;
    const _Float16* p = Bt + (size_t)(col0 + n) * ldk + k0 + hi * 8;
    h8 lo = *(const h8*)(p);
    h8 hv = *(const h8*)(p + 16);
    v16h r;
#pragma unroll
    for (int i = 0; i < 8; ++i) { r[i] = lo[i]; r[8 + i] = hv[i]; }
    return r;
}

__device__ __forceinline__ v8f wmma_f16(v16h a, v16h b, v8f c) {
    return __builtin_amdgcn_wmma_f32_16x16x32_f16(
        false, a, false, b, (short)0, c, false, false);
}

// ---------------------------------------------------------------------------
// Weight convert: W[k][n] f32 (row-major KxN) -> Wt[n][k] f16 (K-major)
// ---------------------------------------------------------------------------
__global__ void conv_transpose_kernel(const float* __restrict__ W,
                                      _Float16* __restrict__ Wt,
                                      int Kd, int Nd) {
    int idx = blockIdx.x * blockDim.x + threadIdx.x;
    if (idx >= Kd * Nd) return;
    int nrow = idx / Kd;
    int k    = idx - nrow * Kd;
    Wt[idx] = (_Float16)W[(size_t)k * Nd + nrow];
}

// ---------------------------------------------------------------------------
// Embedding + positional encoding (f32 residual stream)
// ---------------------------------------------------------------------------
__global__ void embed_kernel(const int* __restrict__ loc,
                             const int* __restrict__ users,
                             const float* __restrict__ loc_emb,
                             const float* __restrict__ user_emb,
                             float* __restrict__ X) {
    int idx = blockIdx.x * blockDim.x + threadIdx.x;
    if (idx >= BS * D) return;
    int t = idx >> 8;          // / D
    int d = idx & (D - 1);
    int b = t >> 7;            // / SEQ
    int s = t & (SEQ - 1);
    int i2 = d & ~1;
    float ang = (float)s * __powf(10000.0f, -(float)i2 / (float)D);
    float pe  = (d & 1) ? __cosf(ang) : __sinf(ang);
    X[idx] = loc_emb[(size_t)loc[t] * D + d] +
             user_emb[(size_t)users[b] * D + d] + pe;
}

// ---------------------------------------------------------------------------
// LayerNorm over D=256, one token per 256-thread block; writes f16
// token index = blockIdx.x * tstride + toff ; output row = blockIdx.x
// ---------------------------------------------------------------------------
__global__ void __launch_bounds__(256) ln_kernel(const float* __restrict__ X,
                                                 const float* __restrict__ g,
                                                 const float* __restrict__ bt,
                                                 _Float16* __restrict__ Hm,
                                                 int tstride, int toff) {
    __shared__ float red[D];
    int t = blockIdx.x * tstride + toff;
    int i = threadIdx.x;
    float v = X[(size_t)t * D + i];
    red[i] = v;
    __syncthreads();
#pragma unroll
    for (int s2 = 128; s2 > 0; s2 >>= 1) {
        if (i < s2) red[i] += red[i + s2];
        __syncthreads();
    }
    float mu = red[0] * (1.0f / D);
    __syncthreads();
    float dv = v - mu;
    red[i] = dv * dv;
    __syncthreads();
#pragma unroll
    for (int s2 = 128; s2 > 0; s2 >>= 1) {
        if (i < s2) red[i] += red[i + s2];
        __syncthreads();
    }
    float var = red[0] * (1.0f / D);
    float y = dv * rsqrtf(var + 1e-5f) * g[i] + bt[i];
    Hm[(size_t)blockIdx.x * D + i] = (_Float16)y;
}

// ---------------------------------------------------------------------------
// Generic WMMA GEMM with 2x2 register tiling:
// each wave computes a 32x32 patch (4 accumulators), reusing A fragments
// across 2 column tiles and B fragments across 2 row tiles
// (8 b128 loads -> 4 WMMAs per k-step).  The column-edge predicate is
// hoisted out of the k-loop so the hot path has no EXEC manipulation.
// Block = 8 waves arranged 4x2 -> 128x64 block tile.
// C[M,N] = act( A[M,K]f16 @ Bt[N,K]f16 + bias ) + residual, optional
// (1 - sigmoid(*wptr)) output scale (final logits).
// ---------------------------------------------------------------------------
__device__ __forceinline__ void store_tile(v8f c, int row0, int col0,
                                           const float* __restrict__ bias,
                                           const float* __restrict__ residual,
                                           float* __restrict__ Cf,
                                           _Float16* __restrict__ Ch,
                                           int relu, float scale, int N) {
    int lane = threadIdx.x & 31;
    int n = lane & 15, hi = lane >> 4;
    int col = col0 + n;
    float bv = bias ? bias[col] : 0.0f;
#pragma unroll
    for (int r = 0; r < 8; ++r) {
        int row = row0 + r + hi * 8;
        size_t idx = (size_t)row * N + col;
        float v = c[r] + bv;
        if (relu) v = fmaxf(v, 0.0f);
        if (residual) v += residual[idx];
        v *= scale;
        if (Cf) Cf[idx] = v;
        if (Ch) Ch[idx] = (_Float16)v;
    }
}

__global__ void __launch_bounds__(256) gemm_wmma_kernel(
    const _Float16* __restrict__ A, const _Float16* __restrict__ Bt,
    int M, int N, int K,
    const float* __restrict__ bias, const float* __restrict__ residual,
    float* __restrict__ Cf, _Float16* __restrict__ Ch,
    int relu, const float* __restrict__ wptr) {
    int wid = threadIdx.x >> 5;
    int r0 = blockIdx.x * 128 + (wid & 3) * 32;   // rows r0, r0+16
    int c0 = blockIdx.y * 64 + (wid >> 2) * 32;   // cols c0, c0+16
    int c1 = c0 + 16;
    if (r0 >= M || c0 >= N) return;               // col tiles fill left-to-right
    bool ok1 = c1 < N;                            // wave-uniform edge predicate
    v8f a00 = {}, a01 = {}, a10 = {}, a11 = {};
    if (ok1) {
        for (int k0 = 0; k0 < K; k0 += 32) {      // hot path: 4 WMMA / k-step
            v16h fa0 = load_a_frag(A, K, r0, k0);
            v16h fa1 = load_a_frag(A, K, r0 + 16, k0);
            v16h fb0 = load_bt_frag(Bt, K, c0, k0);
            v16h fb1 = load_bt_frag(Bt, K, c1, k0);
            a00 = wmma_f16(fa0, fb0, a00);
            a10 = wmma_f16(fa1, fb0, a10);
            a01 = wmma_f16(fa0, fb1, a01);
            a11 = wmma_f16(fa1, fb1, a11);
        }
    } else {
        for (int k0 = 0; k0 < K; k0 += 32) {      // right edge: 2 WMMA / k-step
            v16h fa0 = load_a_frag(A, K, r0, k0);
            v16h fa1 = load_a_frag(A, K, r0 + 16, k0);
            v16h fb0 = load_bt_frag(Bt, K, c0, k0);
            a00 = wmma_f16(fa0, fb0, a00);
            a10 = wmma_f16(fa1, fb0, a10);
        }
    }
    float scale = 1.0f;
    if (wptr) {
        float w = 1.0f / (1.0f + __expf(-wptr[0]));
        scale = 1.0f - w;
    }
    store_tile(a00, r0,      c0, bias, residual, Cf, Ch, relu, scale, N);
    store_tile(a10, r0 + 16, c0, bias, residual, Cf, Ch, relu, scale, N);
    if (ok1) {
        store_tile(a01, r0,      c1, bias, residual, Cf, Ch, relu, scale, N);
        store_tile(a11, r0 + 16, c1, bias, residual, Cf, Ch, relu, scale, N);
    }
}

// ---------------------------------------------------------------------------
// Fused QKV projection: H(BSxD) @ {Wq,Wk,Wv} -> per-head layouts
// Q,K: [b][h][s][dk] ; V stored transposed: [b][h][dk][s]
// Block = 4 waves, each wave owns one 16x16 output tile (reuses A 3x).
// ---------------------------------------------------------------------------
__global__ void __launch_bounds__(128) qkv_kernel(
    const _Float16* __restrict__ Hm,
    const _Float16* __restrict__ Wqt, const _Float16* __restrict__ Wkt,
    const _Float16* __restrict__ Wvt,
    _Float16* __restrict__ Q, _Float16* __restrict__ Kx,
    _Float16* __restrict__ Vt) {
    int wid  = threadIdx.x >> 5;
    int tile = blockIdx.x * 4 + wid;          // 32768 tiles total
    int mt = tile >> 4;                       // D/16 = 16 col tiles
    int nt = tile & 15;
    int row0 = mt * 16, col0 = nt * 16;
    v8f cq = {}, ck = {}, cv = {};
    for (int k0 = 0; k0 < D; k0 += 32) {
        v16h a = load_a_frag(Hm, D, row0, k0);
        cq = wmma_f16(a, load_bt_frag(Wqt, D, col0, k0), cq);
        ck = wmma_f16(a, load_bt_frag(Wkt, D, col0, k0), ck);
        cv = wmma_f16(a, load_bt_frag(Wvt, D, col0, k0), cv);
    }
    int lane = threadIdx.x & 31;
    int n = lane & 15, hi = lane >> 4;
    int col = col0 + n;
    int head = col >> 5, dk = col & (DK - 1);
#pragma unroll
    for (int r = 0; r < 8; ++r) {
        int t = row0 + r + hi * 8;
        int b = t >> 7;            // / SEQ
        int s = t & (SEQ - 1);
        size_t qi = (((size_t)b * NH + head) * SEQ + s) * DK + dk;
        Q[qi]  = (_Float16)cq[r];
        Kx[qi] = (_Float16)ck[r];
        Vt[(((size_t)b * NH + head) * DK + dk) * SEQ + s] = (_Float16)cv[r];
    }
}

// ---------------------------------------------------------------------------
// Attention: one block per (b, head). 8 waves, each owns 16 query rows.
// QK^T: K=DK=32 -> single 16x16x32 WMMA per 16x16 score tile (8 tiles).
// Row softmax via xor-shuffles within 16-lane groups, probs -> LDS (f16),
// then PV: probs(LDS) @ V (Vt K-major) with 4 K-steps.
// ---------------------------------------------------------------------------
__global__ void __launch_bounds__(256) attn_kernel(
    const _Float16* __restrict__ Q, const _Float16* __restrict__ Kx,
    const _Float16* __restrict__ Vt, _Float16* __restrict__ O) {
    __shared__ _Float16 probs[SEQ * SEQ];     // 32 KB
    int bh   = blockIdx.x;                    // b*NH + head
    int wid  = threadIdx.x >> 5;
    int row0 = wid * 16;
    int lane = threadIdx.x & 31;
    int n = lane & 15, hi = lane >> 4;

    const _Float16* qb = Q  + (size_t)bh * SEQ * DK;
    const _Float16* kb = Kx + (size_t)bh * SEQ * DK;

    v16h a = load_a_frag(qb, DK, row0, 0);
    v8f sc[8];
#pragma unroll
    for (int ct = 0; ct < 8; ++ct) {
        v8f z = {};
        sc[ct] = wmma_f16(a, load_bt_frag(kb, DK, ct * 16, 0), z);
    }
    const float scale = 0.17677669529663687f; // 1/sqrt(DK)
#pragma unroll
    for (int r = 0; r < 8; ++r) {
        float m = -1e30f;
#pragma unroll
        for (int ct = 0; ct < 8; ++ct) {
            sc[ct][r] *= scale;
            m = fmaxf(m, sc[ct][r]);
        }
#pragma unroll
        for (int off = 8; off >= 1; off >>= 1)
            m = fmaxf(m, __shfl_xor(m, off, 32));
        float sum = 0.0f;
#pragma unroll
        for (int ct = 0; ct < 8; ++ct) {
            float e = __expf(sc[ct][r] - m);
            sc[ct][r] = e;
            sum += e;
        }
#pragma unroll
        for (int off = 8; off >= 1; off >>= 1)
            sum += __shfl_xor(sum, off, 32);
        float inv = 1.0f / sum;
        int row = row0 + r + hi * 8;
#pragma unroll
        for (int ct = 0; ct < 8; ++ct)
            probs[row * SEQ + ct * 16 + n] = (_Float16)(sc[ct][r] * inv);
    }
    __syncthreads();

    const _Float16* vb = Vt + (size_t)bh * DK * SEQ;   // [dk][s]
    v8f o0 = {}, o1 = {};
    for (int k0 = 0; k0 < SEQ; k0 += 32) {
        v16h ap = load_a_frag(probs, SEQ, row0, k0);   // LDS reads
        o0 = wmma_f16(ap, load_bt_frag(vb, SEQ, 0,  k0), o0);
        o1 = wmma_f16(ap, load_bt_frag(vb, SEQ, 16, k0), o1);
    }
    int b = bh >> 3, head = bh & (NH - 1);
#pragma unroll
    for (int r = 0; r < 8; ++r) {
        int s = row0 + r + hi * 8;
        size_t base = ((size_t)b * SEQ + s) * D + head * DK;
        O[base + n]      = (_Float16)o0[r];
        O[base + 16 + n] = (_Float16)o1[r];
    }
}

// ---------------------------------------------------------------------------
// Return-boost scatter: out[b, recent_j] += sigmoid(w)*rs*boost[j]
// ---------------------------------------------------------------------------
__global__ void boost_kernel(const int* __restrict__ loc,
                             const float* __restrict__ pboost,
                             const float* __restrict__ rstr,
                             const float* __restrict__ ew,
                             float* __restrict__ out) {
    int b = blockIdx.x * blockDim.x + threadIdx.x;
    if (b >= BSZ) return;
    float w  = 1.0f / (1.0f + __expf(-ew[0]));
    float rs = rstr[0];
    for (int j = 0; j < 5; ++j) {
        int r = loc[b * SEQ + (SEQ - 1 - j)];
        out[(size_t)b * VOC + r] += w * rs * pboost[j];
    }
}

// ---------------------------------------------------------------------------
// Host launcher
// ---------------------------------------------------------------------------
static inline int cdiv(long long a, long long b) { return (int)((a + b - 1) / b); }

extern "C" void kernel_launch(void* const* d_in, const int* in_sizes, int n_in,
                              void* d_out, int out_size, void* d_ws, size_t ws_size,
                              hipStream_t stream) {
    const int*   locations = (const int*)  d_in[0];
    const int*   users     = (const int*)  d_in[1];
    const float* loc_emb   = (const float*)d_in[2];
    const float* user_emb  = (const float*)d_in[3];
    const float* Wq   = (const float*)d_in[4];
    const float* Wk   = (const float*)d_in[5];
    const float* Wv   = (const float*)d_in[6];
    const float* Wo   = (const float*)d_in[7];
    const float* ln1g = (const float*)d_in[8];
    const float* ln1b = (const float*)d_in[9];
    const float* ln2g = (const float*)d_in[10];
    const float* ln2b = (const float*)d_in[11];
    const float* W1   = (const float*)d_in[12];
    const float* b1   = (const float*)d_in[13];
    const float* W2   = (const float*)d_in[14];
    const float* b2   = (const float*)d_in[15];
    const float* lnfg = (const float*)d_in[16];
    const float* lnfb = (const float*)d_in[17];
    const float* Wout = (const float*)d_in[18];
    const float* bout = (const float*)d_in[19];
    const float* pboost = (const float*)d_in[20];
    const float* rstr   = (const float*)d_in[21];
    const float* ew     = (const float*)d_in[22];
    (void)in_sizes; (void)n_in; (void)out_size; (void)ws_size;

    // workspace carve (256B aligned)
    char* ws = (char*)d_ws;
    size_t off = 0;
    auto carve = [&](size_t bytes) -> void* {
        void* p = ws + off;
        off += (bytes + 255) & ~(size_t)255;
        return p;
    };
    float*    x    = (float*)   carve((size_t)BS * D * 4);
    _Float16* hbuf = (_Float16*)carve((size_t)BS * D * 2);
    _Float16* qb   = (_Float16*)carve((size_t)BS * D * 2);
    _Float16* kb   = (_Float16*)carve((size_t)BS * D * 2);
    _Float16* vt   = (_Float16*)carve((size_t)BS * D * 2);
    _Float16* ob   = (_Float16*)carve((size_t)BS * D * 2);
    _Float16* act  = (_Float16*)carve((size_t)BS * DI * 2);
    _Float16* xf   = (_Float16*)carve((size_t)BSZ * D * 2);
    _Float16 *wqt[NL], *wkt[NL], *wvt[NL], *wot[NL], *w1t[NL], *w2t[NL];
    for (int l = 0; l < NL; ++l) {
        wqt[l] = (_Float16*)carve((size_t)D * D * 2);
        wkt[l] = (_Float16*)carve((size_t)D * D * 2);
        wvt[l] = (_Float16*)carve((size_t)D * D * 2);
        wot[l] = (_Float16*)carve((size_t)D * D * 2);
        w1t[l] = (_Float16*)carve((size_t)D * DI * 2);
        w2t[l] = (_Float16*)carve((size_t)DI * D * 2);
    }
    _Float16* woutt = (_Float16*)carve((size_t)VOC * D * 2);

    // 1) weight conversion to f16 K-major
    for (int l = 0; l < NL; ++l) {
        conv_transpose_kernel<<<cdiv((long long)D * D, 256), 256, 0, stream>>>(
            Wq + (size_t)l * D * D, wqt[l], D, D);
        conv_transpose_kernel<<<cdiv((long long)D * D, 256), 256, 0, stream>>>(
            Wk + (size_t)l * D * D, wkt[l], D, D);
        conv_transpose_kernel<<<cdiv((long long)D * D, 256), 256, 0, stream>>>(
            Wv + (size_t)l * D * D, wvt[l], D, D);
        conv_transpose_kernel<<<cdiv((long long)D * D, 256), 256, 0, stream>>>(
            Wo + (size_t)l * D * D, wot[l], D, D);
        conv_transpose_kernel<<<cdiv((long long)D * DI, 256), 256, 0, stream>>>(
            W1 + (size_t)l * D * DI, w1t[l], D, DI);
        conv_transpose_kernel<<<cdiv((long long)DI * D, 256), 256, 0, stream>>>(
            W2 + (size_t)l * DI * D, w2t[l], DI, D);
    }
    conv_transpose_kernel<<<cdiv((long long)D * VOC, 256), 256, 0, stream>>>(
        Wout, woutt, D, VOC);

    // 2) embedding + posenc
    embed_kernel<<<cdiv((long long)BS * D, 256), 256, 0, stream>>>(
        locations, users, loc_emb, user_emb, x);

    // 3) transformer layers
    for (int l = 0; l < NL; ++l) {
        ln_kernel<<<BS, 256, 0, stream>>>(x, ln1g + l * D, ln1b + l * D, hbuf, 1, 0);
        qkv_kernel<<<(BS / 16) * (D / 16) / 4, 128, 0, stream>>>(
            hbuf, wqt[l], wkt[l], wvt[l], qb, kb, vt);
        attn_kernel<<<BSZ * NH, 256, 0, stream>>>(qb, kb, vt, ob);
        gemm_wmma_kernel<<<dim3(cdiv(BS, 128), cdiv(D, 64)), 256, 0, stream>>>(
            ob, wot[l], BS, D, D, nullptr, x, x, nullptr, 0, nullptr);
        ln_kernel<<<BS, 256, 0, stream>>>(x, ln2g + l * D, ln2b + l * D, hbuf, 1, 0);
        gemm_wmma_kernel<<<dim3(cdiv(BS, 128), cdiv(DI, 64)), 256, 0, stream>>>(
            hbuf, w1t[l], BS, DI, D, b1 + l * DI, nullptr, nullptr, act, 1, nullptr);
        gemm_wmma_kernel<<<dim3(cdiv(BS, 128), cdiv(D, 64)), 256, 0, stream>>>(
            act, w2t[l], BS, D, DI, b2 + l * D, x, x, nullptr, 0, nullptr);
    }

    // 4) final LN (last token per batch) + output projection with (1-w) scale
    ln_kernel<<<BSZ, 256, 0, stream>>>(x, lnfg, lnfb, xf, SEQ, SEQ - 1);
    gemm_wmma_kernel<<<dim3(cdiv(BSZ, 128), cdiv(VOC, 64)), 256, 0, stream>>>(
        xf, woutt, BSZ, VOC, D, bout, nullptr, (float*)d_out, nullptr, 0, ew);

    // 5) return-location boost
    boost_kernel<<<1, 256, 0, stream>>>(locations, pboost, rstr, ew, (float*)d_out);
}
